// MultiHeadContrastive_34926674051798
// MI455X (gfx1250) — matrix operand order repaired and checked by reading, not compiled, not run on hardware
//
#include <hip/hip_runtime.h>

// ---------------- problem constants ----------------
static constexpr int N    = 8192;
static constexpr int C    = 1024;
static constexpr int HID  = 256;
static constexpr int FGD  = 64;
static constexpr int CLSD = 128;
static constexpr int NTILES = N / 16;      // 512 column tiles
static constexpr int JCHUNKS = 8;          // split j-sweep for occupancy
static constexpr int TILES_PER_CHUNK = NTILES / JCHUNKS;  // 64

typedef __attribute__((ext_vector_type(16))) _Float16 v16h;
typedef __attribute__((ext_vector_type(8)))  _Float16 v8h;
typedef __attribute__((ext_vector_type(8)))  float    v8f;

union FragU { v16h v; v8h h[2]; };

// ---- WMMA fragment loaders (contiguous k = 16*hi + s per lane half for A and B, wave32) ----
// A 16x32 f16 tile from row-major src (rows = M, contiguous K):
//   lane l (m=l&15, hi=l>>4): halves[0..15]=K[k+hi*16 .. +15]
__device__ __forceinline__ v16h loadA16x32(const _Float16* __restrict__ src, int ld,
                                           int mBase, int k) {
    const int l  = threadIdx.x & 31;
    const int m  = l & 15;
    const int hi = l >> 4;
    const _Float16* p = src + (size_t)(mBase + m) * ld + k;
    FragU f;
    f.h[0] = *(const v8h*)(p + hi * 16);
    f.h[1] = *(const v8h*)(p + hi * 16 + 8);
    return f.v;
}

// B 32x16 f16 tile for NT-GEMM (B[k][n] = W[nBase+n][kBase+k], W row-major):
//   lane l (n=l&15, hi=l>>4): halves[0..15] = W[n][k + hi*16 .. +15]
__device__ __forceinline__ v16h loadB32x16(const _Float16* __restrict__ src, int ld,
                                           int nBase, int k) {
    const int l  = threadIdx.x & 31;
    const int n  = l & 15;
    const int hi = l >> 4;
    const _Float16* p = src + (size_t)(nBase + n) * ld + k + hi * 16;
    FragU f;
    f.h[0] = *(const v8h*)(p);
    f.h[1] = *(const v8h*)(p + 8);
    return f.v;
}

__device__ __forceinline__ v8f wmma_f16(v16h a, v16h b, v8f c) {
    return __builtin_amdgcn_wmma_f32_16x16x32_f16(false, a, false, b, (short)0, c,
                                                  false, false);
}

// ---------------- utility kernels ----------------
__global__ void cvt_f32_to_f16(const float* __restrict__ src, _Float16* __restrict__ dst,
                               int n) {
    int i = blockIdx.x * blockDim.x + threadIdx.x;
    if (i < n) dst[i] = (_Float16)src[i];
}

// Integer histogram over labels in {-1..20} -> bins[label+1]; exact & deterministic.
__global__ void hist_kernel(const int* __restrict__ labels, int* __restrict__ hist) {
    __shared__ int h[32];
    if (threadIdx.x < 32) h[threadIdx.x] = 0;
    __syncthreads();
    for (int i = threadIdx.x; i < N; i += blockDim.x) {
        atomicAdd(&h[labels[i] + 1], 1);
    }
    __syncthreads();
    if (threadIdx.x < 32) hist[threadIdx.x] = h[threadIdx.x];
}

// ---------------- GEMM1: H = relu(X @ W1^T + b1), f16 out ----------------
// grid: (N/16, HID/16), block: 32 (one wave per 16x16 tile), K = C = 1024
__global__ void gemm1_relu(const _Float16* __restrict__ X, const _Float16* __restrict__ W,
                           const float* __restrict__ bias, _Float16* __restrict__ H) {
    const int mBase = blockIdx.x * 16;
    const int nBase = blockIdx.y * 16;
    const int l  = threadIdx.x & 31;
    const int hi = l >> 4;
    const int n  = nBase + (l & 15);
    v8f acc = {};
    for (int k = 0; k < C; k += 32) {
        v16h a = loadA16x32(X, C, mBase, k);
        v16h b = loadB32x16(W, C, nBase, k);
        acc = wmma_f16(a, b, acc);
    }
    const float bv = bias[n];
#pragma unroll
    for (int i = 0; i < 8; ++i) {
        const int row = mBase + i + hi * 8;
        float v = acc[i] + bv;
        H[(size_t)row * HID + n] = (_Float16)(v > 0.0f ? v : 0.0f);
    }
}

// ---------------- GEMM2 + L2 normalize: Z = normalize(H @ W2^T + b2) ----------------
// One wave owns a full 16-row block across all D columns (D/16 tiles in registers).
template <int D>
__global__ void gemm2_norm(const _Float16* __restrict__ H, const _Float16* __restrict__ W2,
                           const float* __restrict__ b2, _Float16* __restrict__ Z) {
    constexpr int NT = D / 16;
    const int mBase = blockIdx.x * 16;
    const int l    = threadIdx.x & 31;
    const int hi   = l >> 4;
    const int nsub = l & 15;

    v8f zero = {};
    v8f acc[NT];
#pragma unroll
    for (int t = 0; t < NT; ++t) acc[t] = zero;

    for (int k = 0; k < HID; k += 32) {
        v16h a = loadA16x32(H, HID, mBase, k);
#pragma unroll
        for (int t = 0; t < NT; ++t) {
            v16h b = loadB32x16(W2, HID, t * 16, k);
            acc[t] = wmma_f16(a, b, acc[t]);
        }
    }

    float ss[8];
#pragma unroll
    for (int i = 0; i < 8; ++i) ss[i] = 0.0f;
#pragma unroll
    for (int t = 0; t < NT; ++t) {
        const float bv = b2[t * 16 + nsub];
#pragma unroll
        for (int i = 0; i < 8; ++i) {
            acc[t][i] += bv;
            ss[i] += acc[t][i] * acc[t][i];
        }
    }
    // reduce sum-of-squares across the 16 lanes of each half (rows stay in-half)
#pragma unroll
    for (int m = 8; m >= 1; m >>= 1) {
#pragma unroll
        for (int i = 0; i < 8; ++i) ss[i] += __shfl_xor(ss[i], m, 32);
    }
    float inv[8];
#pragma unroll
    for (int i = 0; i < 8; ++i) inv[i] = 1.0f / fmaxf(sqrtf(ss[i]), 1e-8f);

#pragma unroll
    for (int t = 0; t < NT; ++t) {
#pragma unroll
        for (int i = 0; i < 8; ++i) {
            const int row = mBase + i + hi * 8;
            Z[(size_t)row * D + t * 16 + nsub] = (_Float16)(acc[t][i] * inv[i]);
        }
    }
}

// ---------------- Fused sim + contrastive row statistics ----------------
// grid: (N/16, JCHUNKS); block 32. Anchor A-fragments stay in registers for the
// whole j sweep. outA/outB are [JCHUNKS][N] partial slots (no atomics).
// fg head  (CLSHEAD=false): outA = sum_{j!=i} exp(sim), outB = sum_{pos} exp(sim)
// cls head (CLSHEAD=true ): outA = sum_{j!=i,!ign} exp(sim), outB = sum_{pos} sim
template <int D, bool CLSHEAD>
__global__ void sim_stats(const _Float16* __restrict__ Z, const int* __restrict__ labels,
                          float* __restrict__ outA, float* __restrict__ outB) {
    constexpr int KST = D / 32;
    const int mBase = blockIdx.x * 16;
    const int chunk = blockIdx.y;
    const int l    = threadIdx.x & 31;
    const int hi   = l >> 4;
    const int nsub = l & 15;

    v16h afrag[KST];
#pragma unroll
    for (int s = 0; s < KST; ++s) afrag[s] = loadA16x32(Z, D, mBase, s * 32);

    int labR[8], rowG[8];
#pragma unroll
    for (int i = 0; i < 8; ++i) {
        rowG[i] = mBase + i + hi * 8;
        labR[i] = labels[rowG[i]];
    }

    float accA[8], accB[8];
#pragma unroll
    for (int i = 0; i < 8; ++i) { accA[i] = 0.0f; accB[i] = 0.0f; }

    const int jtBeg = chunk * TILES_PER_CHUNK;
    const int jtEnd = jtBeg + TILES_PER_CHUNK;
    for (int jt = jtBeg; jt < jtEnd; ++jt) {
        v8f sim = {};
#pragma unroll
        for (int s = 0; s < KST; ++s) {
            v16h b = loadB32x16(Z, D, jt * 16, s * 32);
            sim = wmma_f16(afrag[s], b, sim);
        }
        if (jt + 1 < jtEnd) {
            __builtin_prefetch(Z + (size_t)((jt + 1) * 16 + nsub) * D, 0, 0);
        }
        const int  j    = jt * 16 + nsub;
        const int  labJ = labels[j];
        const bool ignJ = labJ < 0;
        const bool fgJ  = labJ > 0;
#pragma unroll
        for (int i = 0; i < 8; ++i) {
            const float s5 = sim[i] * 5.0f;  // 1/tau = 5
            const float e  = __expf(s5);
            const bool notSelf = (j != rowG[i]);
            if (CLSHEAD) {
                if (notSelf && !ignJ) accA[i] += e;
                if (notSelf && fgJ && labJ == labR[i]) accB[i] += s5;
            } else {
                if (notSelf) accA[i] += e;
                if (notSelf && fgJ) accB[i] += e;
            }
        }
    }

#pragma unroll
    for (int m = 8; m >= 1; m >>= 1) {
#pragma unroll
        for (int i = 0; i < 8; ++i) {
            accA[i] += __shfl_xor(accA[i], m, 32);
            accB[i] += __shfl_xor(accB[i], m, 32);
        }
    }
    if (nsub == 0) {  // lanes 0 (rows 0..7) and 16 (rows 8..15) write
#pragma unroll
        for (int i = 0; i < 8; ++i) {
            const int r = rowG[i];
            outA[(size_t)chunk * N + r] = accA[i];
            outB[(size_t)chunk * N + r] = accB[i];
        }
    }
}

// ---------------- finalize: per-row losses -> weighted means ----------------
__global__ void finalize_kernel(const float* __restrict__ fgDen, const float* __restrict__ fgNum,
                                const float* __restrict__ clsDen, const float* __restrict__ clsPos,
                                const int* __restrict__ labels, const float* __restrict__ ious,
                                const int* __restrict__ hist, float* __restrict__ out) {
    __shared__ float red[4][256];
    const int t = threadIdx.x;

    int P = 0;                 // #(fg & ~ignore)
    for (int c = 2; c < 22; ++c) P += hist[c];
    int cntNotIgn = 0;         // #(~ignore)
    for (int c = 1; c < 22; ++c) cntNotIgn += hist[c];

    float nF = 0.0f, dF = 0.0f, nC = 0.0f, dC = 0.0f;
    for (int r = t; r < N; r += blockDim.x) {
        const int   L  = labels[r];
        const float io = ious[r];
        const float w  = (io > 0.5f) ? io : 0.0f;
        const bool anchor = (L > 0);

        float den = 0.0f, num = 0.0f, cden = 0.0f, cpos = 0.0f;
        for (int c2 = 0; c2 < JCHUNKS; ++c2) {
            den  += fgDen[(size_t)c2 * N + r];
            num  += fgNum[(size_t)c2 * N + r];
            cden += clsDen[(size_t)c2 * N + r];
            cpos += clsPos[(size_t)c2 * N + r];
        }

        // fg/bg loss
        const int nPosFG = P - (L > 0 ? 1 : 0);
        if (anchor && nPosFG > 0) {
            const float li = -logf((num + 1e-8f) / (den + 1e-8f));
            nF += li * w;
            dF += w;
        }
        // class supcon loss
        const int  nPos = (L > 0) ? (hist[L + 1] - 1) : 0;
        const bool anyValid = (cntNotIgn - (L >= 0 ? 1 : 0)) > 0;
        if (anchor && nPos > 0 && anyValid) {
            const float dl = logf(cden);  // == logsumexp over valid (masked = -1e9)
            const float li = -(cpos - (float)nPos * dl) / (float)nPos;
            nC += li * w;
            dC += w;
        }
    }
    red[0][t] = nF; red[1][t] = dF; red[2][t] = nC; red[3][t] = dC;
    __syncthreads();
    for (int s = 128; s > 0; s >>= 1) {
        if (t < s) {
#pragma unroll
            for (int q = 0; q < 4; ++q) red[q][t] += red[q][t + s];
        }
        __syncthreads();
    }
    if (t == 0) {
        out[0] = red[0][0] / (red[1][0] + 1e-8f);
        out[1] = red[2][0] / (red[3][0] + 1e-12f);
    }
}

// ---------------- host launcher ----------------
extern "C" void kernel_launch(void* const* d_in, const int* in_sizes, int n_in,
                              void* d_out, int out_size, void* d_ws, size_t ws_size,
                              hipStream_t stream) {
    const float* X    = (const float*)d_in[0];
    const int*   lab  = (const int*)d_in[1];
    const float* ious = (const float*)d_in[2];
    const float* w1f  = (const float*)d_in[3];
    const float* b1f  = (const float*)d_in[4];
    const float* w2f  = (const float*)d_in[5];
    const float* b2f  = (const float*)d_in[6];
    const float* w1c  = (const float*)d_in[7];
    const float* b1c  = (const float*)d_in[8];
    const float* w2c  = (const float*)d_in[9];
    const float* b2c  = (const float*)d_in[10];
    float* out = (float*)d_out;

    // workspace carve-up (256B aligned)
    char* ws = (char*)d_ws;
    size_t off = 0;
    auto alloc = [&](size_t bytes) {
        size_t o = off;
        off = (off + bytes + 255) & ~(size_t)255;
        return o;
    };
    _Float16* Xh   = (_Float16*)(ws + alloc((size_t)N * C * 2));
    _Float16* W1fh = (_Float16*)(ws + alloc((size_t)HID * C * 2));
    _Float16* W1ch = (_Float16*)(ws + alloc((size_t)HID * C * 2));
    _Float16* W2fh = (_Float16*)(ws + alloc((size_t)FGD * HID * 2));
    _Float16* W2ch = (_Float16*)(ws + alloc((size_t)CLSD * HID * 2));
    _Float16* Hf   = (_Float16*)(ws + alloc((size_t)N * HID * 2));
    _Float16* Hc   = (_Float16*)(ws + alloc((size_t)N * HID * 2));
    _Float16* Zf   = (_Float16*)(ws + alloc((size_t)N * FGD * 2));
    _Float16* Zc   = (_Float16*)(ws + alloc((size_t)N * CLSD * 2));
    float* fgDenP  = (float*)(ws + alloc((size_t)JCHUNKS * N * 4));
    float* fgNumP  = (float*)(ws + alloc((size_t)JCHUNKS * N * 4));
    float* clsDenP = (float*)(ws + alloc((size_t)JCHUNKS * N * 4));
    float* clsPosP = (float*)(ws + alloc((size_t)JCHUNKS * N * 4));
    int*   hist    = (int*)(ws + alloc(32 * 4));

    // 1) f32 -> f16 conversions
    auto cvt = [&](const float* s, _Float16* d, int n) {
        cvt_f32_to_f16<<<(n + 255) / 256, 256, 0, stream>>>(s, d, n);
    };
    cvt(X,   Xh,   N * C);
    cvt(w1f, W1fh, HID * C);
    cvt(w1c, W1ch, HID * C);
    cvt(w2f, W2fh, FGD * HID);
    cvt(w2c, W2ch, CLSD * HID);

    // 2) label histogram (exact integer)
    hist_kernel<<<1, 256, 0, stream>>>(lab, hist);

    // 3) GEMM1 + ReLU (both heads): wave-per-tile WMMA NT-GEMM
    dim3 g1(N / 16, HID / 16);
    gemm1_relu<<<g1, 32, 0, stream>>>(Xh, W1fh, b1f, Hf);
    gemm1_relu<<<g1, 32, 0, stream>>>(Xh, W1ch, b1c, Hc);

    // 4) GEMM2 + bias + L2-normalize
    gemm2_norm<FGD><<<N / 16, 32, 0, stream>>>(Hf, W2fh, b2f, Zf);
    gemm2_norm<CLSD><<<N / 16, 32, 0, stream>>>(Hc, W2ch, b2c, Zc);

    // 5) fused similarity + masked row statistics
    dim3 gs(N / 16, JCHUNKS);
    sim_stats<FGD,  false><<<gs, 32, 0, stream>>>(Zf, lab, fgDenP,  fgNumP);
    sim_stats<CLSD, true ><<<gs, 32, 0, stream>>>(Zc, lab, clsDenP, clsPosP);

    // 6) finalize to the two scalar losses
    finalize_kernel<<<1, 256, 0, stream>>>(fgDenP, fgNumP, clsDenP, clsPosP,
                                           lab, ious, hist, out);
    (void)in_sizes; (void)n_in; (void)out_size; (void)ws_size;
}
